// MMLinear_25254407700650
// MI455X (gfx1250) — matrix operand order repaired and checked by reading, not compiled, not run on hardware
//
#include <hip/hip_runtime.h>
#include <hip/hip_bf16.h>

// ---------------------------------------------------------------------------
// MoE top-2 (8 experts), T=32768 tokens, L=O=1024. Sparse grouped-GEMM design:
//   1) routing kernel: fp32 logits/softmax/top2, gather per-expert token lists,
//      and (if workspace allows) emit a packed-bf16 copy of x as a side product
//   2) one-shot expert_w fp32->bf16 convert kernel (weights stay L2-resident)
//   3) grouped GEMM per expert with BF16 WMMA (v_wmma_f32_16x16x32_bf16),
//      128x128 tiles, double-buffered LDS, pure ld->ds->wmma inner loop,
//      atomicAdd scatter combine with bias+routing-weight folded in.
// Fallback GEMM (converts fp32->bf16 in-kernel) is used when ws is small.
// ---------------------------------------------------------------------------

typedef __attribute__((ext_vector_type(16))) __bf16 v16bf;
typedef __attribute__((ext_vector_type(8)))  float  v8f;

#define N_EXPERTS 8
#define L_DIM     1024
#define O_DIM     1024

union ABFrag {
    v16bf    v;
    unsigned u[8];
};

// pack two fp32 -> two bf16 (RNE)
static __device__ __forceinline__ unsigned pk_bf16(float a, float b) {
#if __has_builtin(__builtin_amdgcn_cvt_pk_bf16_f32)
    auto r = __builtin_amdgcn_cvt_pk_bf16_f32(a, b);
    return __builtin_bit_cast(unsigned, r);
#else
    unsigned ua = __float_as_uint(a);
    unsigned ub = __float_as_uint(b);
    ua += 0x7FFFu + ((ua >> 16) & 1u);
    ub += 0x7FFFu + ((ub >> 16) & 1u);
    return (ua >> 16) | (ub & 0xFFFF0000u);
#endif
}

static __device__ __forceinline__ void pk_store4(unsigned* dst, float4 f0, float4 f1,
                                                 float4 f2, float4 f3) {
    dst[0] = pk_bf16(f0.x, f0.y); dst[1] = pk_bf16(f0.z, f0.w);
    dst[2] = pk_bf16(f1.x, f1.y); dst[3] = pk_bf16(f1.z, f1.w);
    dst[4] = pk_bf16(f2.x, f2.y); dst[5] = pk_bf16(f2.z, f2.w);
    dst[6] = pk_bf16(f3.x, f3.y); dst[7] = pk_bf16(f3.z, f3.w);
}

// ---------------------------------------------------------------------------
// Kernel 1: routing (one thread per token, gate_w in LDS) + optional x->bf16.
// ---------------------------------------------------------------------------
__global__ void moe_route(const float* __restrict__ x,
                          const float* __restrict__ gate_w,
                          int*      __restrict__ counts,
                          int*      __restrict__ tok_list,
                          float*    __restrict__ wt_list,
                          unsigned* __restrict__ xbf,     // [T][L/2] bf16x2, may be null
                          int T)
{
    __shared__ __align__(16) float gsh[N_EXPERTS * L_DIM]; // 32 KB

    for (int i = threadIdx.x; i < N_EXPERTS * L_DIM; i += blockDim.x)
        gsh[i] = gate_w[i];
    __syncthreads();

    const int t = blockIdx.x * blockDim.x + threadIdx.x;
    if (t >= T) return;

    const float4* xr = (const float4*)(x + (size_t)t * L_DIM);
    const float4* gr = (const float4*)gsh;
    uint2* xout = xbf ? (uint2*)(xbf + (size_t)t * (L_DIM / 2)) : nullptr;

    float acc[N_EXPERTS];
#pragma unroll
    for (int e = 0; e < N_EXPERTS; ++e) acc[e] = 0.f;

    for (int k = 0; k < L_DIM / 4; ++k) {
        float4 xv = xr[k];
        if (xout) {
            uint2 pr; pr.x = pk_bf16(xv.x, xv.y); pr.y = pk_bf16(xv.z, xv.w);
            xout[k] = pr;
        }
#pragma unroll
        for (int e = 0; e < N_EXPERTS; ++e) {
            float4 gv = gr[e * (L_DIM / 4) + k];
            acc[e] += xv.x * gv.x + xv.y * gv.y + xv.z * gv.z + xv.w * gv.w;
        }
    }

    // softmax (fp32, matches reference)
    float m = acc[0];
#pragma unroll
    for (int e = 1; e < N_EXPERTS; ++e) m = fmaxf(m, acc[e]);
    float p[N_EXPERTS], s = 0.f;
#pragma unroll
    for (int e = 0; e < N_EXPERTS; ++e) { p[e] = __expf(acc[e] - m); s += p[e]; }
    const float inv = 1.f / s;

    // top-1 then top-2 (strict '>' => lowest index wins ties, like lax.top_k)
    int e1 = 0; float b1 = p[0];
#pragma unroll
    for (int e = 1; e < N_EXPERTS; ++e) if (p[e] > b1) { b1 = p[e]; e1 = e; }
    int e2 = (e1 == 0) ? 1 : 0; float b2 = p[e2];
#pragma unroll
    for (int e = 0; e < N_EXPERTS; ++e)
        if (e != e1 && p[e] > b2) { b2 = p[e]; e2 = e; }

    int pos1 = atomicAdd(&counts[e1], 1);
    tok_list[(size_t)e1 * T + pos1] = t;
    wt_list [(size_t)e1 * T + pos1] = b1 * inv;

    int pos2 = atomicAdd(&counts[e2], 1);
    tok_list[(size_t)e2 * T + pos2] = t;
    wt_list [(size_t)e2 * T + pos2] = b2 * inv;
}

// ---------------------------------------------------------------------------
// Kernel 2: expert_w fp32 -> packed bf16 (one shot, fully coalesced b128/b128).
// Each thread: 8 floats -> 4 dwords.
// ---------------------------------------------------------------------------
__global__ void conv_w_bf16(const float* __restrict__ w, unsigned* __restrict__ wbf,
                            int n_oct)   // n_oct = total_elems / 8
{
    int i = blockIdx.x * blockDim.x + threadIdx.x;
    if (i >= n_oct) return;
    const float4* s = (const float4*)w + 2 * (size_t)i;
    float4 a = s[0], b = s[1];
    uint4 o;
    o.x = pk_bf16(a.x, a.y); o.y = pk_bf16(a.z, a.w);
    o.z = pk_bf16(b.x, b.y); o.w = pk_bf16(b.z, b.w);
    ((uint4*)wbf)[i] = o;
}

// ---------------------------------------------------------------------------
// Shared GEMM skeleton pieces
// ---------------------------------------------------------------------------
#define GEMM_PROLOG(EXTRA)                                                        \
    const int e   = blockIdx.z;                                                   \
    const int cnt = counts[e];                                                    \
    const int m0  = blockIdx.x * 128;                                             \
    if (m0 >= cnt) return;                                                        \
    const int n0  = blockIdx.y * 128;                                             \
    __shared__ unsigned Ash[2][128 * 16];                                         \
    __shared__ unsigned Bsh[2][128 * 16];                                         \
    __shared__ int      tok_sh[128];                                              \
    __shared__ float    wgt_sh[128];                                              \
    const int tid = threadIdx.x;                                                  \
    if (tid < 128) {                                                              \
        int g = m0 + tid;                                                         \
        if (g < cnt) {                                                            \
            tok_sh[tid] = tok_list[(size_t)e * T + g];                            \
            wgt_sh[tid] = wt_list [(size_t)e * T + g];                            \
        } else { tok_sh[tid] = 0; wgt_sh[tid] = 0.f; }                            \
    }                                                                             \
    __syncthreads();                                                              \
    const int lane = tid & 31;                                                    \
    const int wv   = tid >> 5;                                                    \
    const int wm   = wv & 3;                                                      \
    const int wn   = wv >> 2;                                                     \
    v8f acc[2][4];                                                                \
    _Pragma("unroll") for (int i = 0; i < 2; ++i)                                 \
        _Pragma("unroll") for (int j = 0; j < 4; ++j)                             \
            acc[i][j] = (v8f){0,0,0,0,0,0,0,0};                                   \
    const int arow  = tid >> 1;                                                   \
    const int ahalf = (tid & 1) * 16;                                             \
    (void)0

// fragment reads + 8 WMMAs on current buffer
static __device__ __forceinline__ void gemm_math(const unsigned* Abuf, const unsigned* Bbuf,
                                                 int lane, int wm, int wn, v8f acc[2][4]) {
    ABFrag afr[2];
    const int koff = (lane >> 4) << 2;
#pragma unroll
    for (int ms = 0; ms < 2; ++ms) {
        const unsigned* ap = &Abuf[(wm * 32 + ms * 16 + (lane & 15)) * 16];
#pragma unroll
        for (int v = 0; v < 8; ++v)
            afr[ms].u[v] = ap[(koff << 1) + v];
    }
#pragma unroll
    for (int ns = 0; ns < 4; ++ns) {
        const unsigned* bp =
            &Bbuf[(wn * 64 + ns * 16 + (lane & 15)) * 16 + ((lane >> 4) << 3)];
        ABFrag bfr;
#pragma unroll
        for (int v = 0; v < 8; ++v) bfr.u[v] = bp[v];
        acc[0][ns] = __builtin_amdgcn_wmma_f32_16x16x32_bf16(
            false, afr[0].v, false, bfr.v, (short)0, acc[0][ns], false, false);
        acc[1][ns] = __builtin_amdgcn_wmma_f32_16x16x32_bf16(
            false, afr[1].v, false, bfr.v, (short)0, acc[1][ns], false, false);
    }
}

static __device__ __forceinline__ void gemm_epilog(const float* expert_b, float* out,
                                                   const int* tok_sh, const float* wgt_sh,
                                                   v8f acc[2][4], int e, int n0, int m0,
                                                   int cnt, int lane, int wm, int wn) {
    const int col   = lane & 15;
    const int rhalf = (lane >> 4) * 8;
#pragma unroll
    for (int ns = 0; ns < 4; ++ns) {
        const int o = n0 + wn * 64 + ns * 16 + col;
        const float bias = expert_b[(size_t)e * O_DIM + o];
#pragma unroll
        for (int ms = 0; ms < 2; ++ms) {
#pragma unroll
            for (int v = 0; v < 8; ++v) {
                const int mrow = wm * 32 + ms * 16 + v + rhalf;
                if (m0 + mrow < cnt) {
                    const float val = wgt_sh[mrow] * (acc[ms][ns][v] + bias);
                    atomicAdd(&out[(size_t)tok_sh[mrow] * O_DIM + o], val);
                }
            }
        }
    }
}

// ---------------------------------------------------------------------------
// Kernel 3a: grouped GEMM over pre-converted bf16 operands.
// Inner loop is pure global_load_b128 -> ds_store_b128 -> ds_load_b128 -> wmma.
// ---------------------------------------------------------------------------
__global__ void __launch_bounds__(256)
moe_gemm_pre(const unsigned* __restrict__ xbf,   // [T][L/2] bf16x2
             const unsigned* __restrict__ wbf,   // [E][O][L/2] bf16x2
             const float*    __restrict__ expert_b,
             const int*      __restrict__ counts,
             const int*      __restrict__ tok_list,
             const float*    __restrict__ wt_list,
             float*          __restrict__ out,
             int T)
{
    GEMM_PROLOG();

    const unsigned* xrow = xbf + (size_t)tok_sh[arow] * (L_DIM / 2) + (ahalf >> 1);
    const unsigned* brow = wbf + ((size_t)e * O_DIM + (n0 + arow)) * (L_DIM / 2) + (ahalf >> 1);
    unsigned* const Adst = &Ash[0][arow * 16 + (ahalf >> 1)];
    unsigned* const Bdst = &Bsh[0][arow * 16 + (ahalf >> 1)];
    const int bufstride = 128 * 16;

    // prologue: stage K-block 0 into buffer 0 (no conversion, straight copy)
    {
        const uint4* sa = (const uint4*)xrow;
        const uint4* sb = (const uint4*)brow;
        ((uint4*)Adst)[0] = sa[0]; ((uint4*)Adst)[1] = sa[1];
        ((uint4*)Bdst)[0] = sb[0]; ((uint4*)Bdst)[1] = sb[1];
    }
    __syncthreads();

    int cur = 0;
    for (int kb = 0; kb < L_DIM; kb += 32) {
        const bool have_next = (kb + 32) < L_DIM;
        uint4 na0, na1, nb0, nb1;
        if (have_next) {
            const uint4* sa = (const uint4*)(xrow + ((kb + 32) >> 1));
            const uint4* sb = (const uint4*)(brow + ((kb + 32) >> 1));
            na0 = sa[0]; na1 = sa[1];
            nb0 = sb[0]; nb1 = sb[1];
        }

        gemm_math(&Ash[0][cur * bufstride], &Bsh[0][cur * bufstride],
                  lane, wm, wn, acc);

        if (have_next) {
            const int nxt = bufstride - cur * bufstride;
            ((uint4*)(Adst + nxt))[0] = na0; ((uint4*)(Adst + nxt))[1] = na1;
            ((uint4*)(Bdst + nxt))[0] = nb0; ((uint4*)(Bdst + nxt))[1] = nb1;
            __syncthreads();
            cur ^= 1;
        }
    }

    gemm_epilog(expert_b, out, tok_sh, wgt_sh, acc, e, n0, m0, cnt, lane, wm, wn);
}

// ---------------------------------------------------------------------------
// Kernel 3b: fallback GEMM over fp32 operands (converts in-kernel).
// ---------------------------------------------------------------------------
__global__ void __launch_bounds__(256)
moe_gemm_cvt(const float* __restrict__ x,
             const float* __restrict__ expert_w,
             const float* __restrict__ expert_b,
             const int*   __restrict__ counts,
             const int*   __restrict__ tok_list,
             const float* __restrict__ wt_list,
             float*       __restrict__ out,
             int T)
{
    GEMM_PROLOG();

    const float* xrow = x + (size_t)tok_sh[arow] * L_DIM + ahalf;
    const float* brow = expert_w + ((size_t)e * O_DIM + (n0 + arow)) * (size_t)L_DIM + ahalf;
    unsigned* const Adst = &Ash[0][arow * 16 + (ahalf >> 1)];
    unsigned* const Bdst = &Bsh[0][arow * 16 + (ahalf >> 1)];
    const int bufstride = 128 * 16;

    {
        const float4* sa = (const float4*)xrow;
        const float4* sb = (const float4*)brow;
        pk_store4(Adst, sa[0], sa[1], sa[2], sa[3]);
        pk_store4(Bdst, sb[0], sb[1], sb[2], sb[3]);
    }
    __syncthreads();

    int cur = 0;
    for (int kb = 0; kb < L_DIM; kb += 32) {
        const bool have_next = (kb + 32) < L_DIM;
        float4 na0, na1, na2, na3, nb0, nb1, nb2, nb3;
        if (have_next) {
            const float4* sa = (const float4*)(xrow + kb + 32);
            const float4* sb = (const float4*)(brow + kb + 32);
            na0 = sa[0]; na1 = sa[1]; na2 = sa[2]; na3 = sa[3];
            nb0 = sb[0]; nb1 = sb[1]; nb2 = sb[2]; nb3 = sb[3];
        }

        gemm_math(&Ash[0][cur * bufstride], &Bsh[0][cur * bufstride],
                  lane, wm, wn, acc);

        if (have_next) {
            const int nxt = bufstride - cur * bufstride;
            pk_store4(Adst + nxt, na0, na1, na2, na3);
            pk_store4(Bdst + nxt, nb0, nb1, nb2, nb3);
            __syncthreads();
            cur ^= 1;
        }
    }

    gemm_epilog(expert_b, out, tok_sh, wgt_sh, acc, e, n0, m0, cnt, lane, wm, wn);
}

// ---------------------------------------------------------------------------
extern "C" void kernel_launch(void* const* d_in, const int* in_sizes, int n_in,
                              void* d_out, int out_size, void* d_ws, size_t ws_size,
                              hipStream_t stream)
{
    const float* x        = (const float*)d_in[0];  // [T, L]
    const float* gate_w   = (const float*)d_in[1];  // [E, L]
    const float* expert_w = (const float*)d_in[2];  // [E, O, L]
    const float* expert_b = (const float*)d_in[3];  // [E, O]
    float*       out      = (float*)d_out;

    const int T = in_sizes[0] / L_DIM;              // 32768

    // workspace layout:
    //   [counts: 128 B | tok_list: E*T i32 | wt_list: E*T f32 | xbf | wbf]
    const size_t lists_off = 128;
    const size_t xbf_off   = lists_off + (size_t)N_EXPERTS * T * 8;
    const size_t wbf_off   = xbf_off + (size_t)T * L_DIM * 2;
    const size_t need_pre  = wbf_off + (size_t)N_EXPERTS * O_DIM * L_DIM * 2;

    int*      counts   = (int*)d_ws;
    int*      tok_list = (int*)((char*)d_ws + lists_off);
    float*    wt_list  = (float*)((char*)d_ws + lists_off + (size_t)N_EXPERTS * T * 4);
    unsigned* xbf      = (unsigned*)((char*)d_ws + xbf_off);
    unsigned* wbf      = (unsigned*)((char*)d_ws + wbf_off);

    const bool use_pre = (ws_size >= need_pre);

    hipMemsetAsync(d_ws, 0, 128, stream);
    hipMemsetAsync(d_out, 0, (size_t)out_size * sizeof(float), stream);

    // 1) routing + gather (+ x bf16 conversion on the pre path)
    moe_route<<<(T + 255) / 256, 256, 0, stream>>>(
        x, gate_w, counts, tok_list, wt_list, use_pre ? xbf : nullptr, T);

    dim3 grid((T + 127) / 128, O_DIM / 128, N_EXPERTS);
    if (use_pre) {
        // 2) weights fp32 -> bf16 (one shot; stays L2-resident for the GEMM)
        const int n_oct = N_EXPERTS * O_DIM * L_DIM / 8;
        conv_w_bf16<<<(n_oct + 255) / 256, 256, 0, stream>>>(expert_w, wbf, n_oct);
        // 3) grouped expert GEMM on bf16 operands
        moe_gemm_pre<<<grid, 256, 0, stream>>>(xbf, wbf, expert_b, counts,
                                               tok_list, wt_list, out, T);
    } else {
        moe_gemm_cvt<<<grid, 256, 0, stream>>>(x, expert_w, expert_b, counts,
                                               tok_list, wt_list, out, T);
    }
}
